// DeformableValueAttention_36009005809699
// MI455X (gfx1250) — compile-verified, matrix operand order checked by reading, not run on hardware
//
#include <hip/hip_runtime.h>
#include <hip/hip_bf16.h>

// ---------------- problem constants ----------------
#define BATCH 4
#define CDIM  512
#define HGT   32
#define WID   32
#define NQ    1024            // H*W
#define NH    8
#define HD    64              // CDIM / NH

typedef __attribute__((ext_vector_type(16))) __bf16 v16bf;
typedef __attribute__((ext_vector_type(8)))  float  v8f;
typedef int v4i __attribute__((vector_size(16)));

union Frag { v16bf v; unsigned int u[8]; unsigned short s[16]; };
union Acc  { v8f v; float f[8]; };

// gfx1250 async global->LDS DMA (ASYNCcnt-tracked), if the toolchain declares it
#if defined(__AMDGCN__) && __has_builtin(__builtin_amdgcn_global_load_async_to_lds_b128) && __has_builtin(__builtin_amdgcn_s_wait_asynccnt)
#define HAVE_ASYNC 1
#endif

#ifdef HAVE_ASYNC
__device__ __forceinline__ void cp16_async(const void* g, void* l) {
  // generic LDS pointer: low 32 bits are the LDS offset (ISA aperture rules)
  __builtin_amdgcn_global_load_async_to_lds_b128(
      (__attribute__((address_space(1))) v4i*)(unsigned long long)g,
      (__attribute__((address_space(3))) v4i*)(unsigned int)(unsigned long long)l,
      0, 0);
}
#endif

__device__ __forceinline__ unsigned short f2bf(float f) {
  unsigned int u = __float_as_uint(f);
  unsigned int r = u + 0x7FFFu + ((u >> 16) & 1u);   // round-to-nearest-even
  return (unsigned short)(r >> 16);
}

// 16x32 bf16 WMMA fragment (ISA 7.12.2): per lane two contiguous 16-byte runs
// at col offsets {half*8, 16+half*8} -> two b128 loads (global or ds).
__device__ __forceinline__ void load_frag(Frag& f, const unsigned short* p,
                                          int stride, int row, int colbase, int hlf) {
  const unsigned short* base = p + row * stride + colbase + hlf * 8;
  uint4 lo = *(const uint4*)(base);
  uint4 hi = *(const uint4*)(base + 16);
  f.u[0] = lo.x; f.u[1] = lo.y; f.u[2] = lo.z; f.u[3] = lo.w;
  f.u[4] = hi.x; f.u[5] = hi.y; f.u[6] = hi.z; f.u[7] = hi.w;
}

// ---------------- packing kernels ----------------
__global__ __launch_bounds__(256)
void pack_feat(const float* __restrict__ qf, const float* __restrict__ kf,
               unsigned short* __restrict__ Xq, unsigned short* __restrict__ Xk) {
  int idx = blockIdx.x * blockDim.x + threadIdx.x;        // [b][n][c]
  if (idx >= BATCH * NQ * CDIM) return;
  int c = idx & (CDIM - 1);
  int n = (idx >> 9) & (NQ - 1);
  int b = idx >> 19;
  int src = (b * CDIM + c) * NQ + n;                      // [b][c][n]
  Xq[idx] = f2bf(qf[src]);
  Xk[idx] = f2bf(kf[src]);
}

__global__ __launch_bounds__(256)
void pack_w(const float* __restrict__ Wq, const float* __restrict__ Wk,
            const float* __restrict__ Wv, const float* __restrict__ Wo,
            unsigned short* __restrict__ oq, unsigned short* __restrict__ ok,
            unsigned short* __restrict__ ov, unsigned short* __restrict__ oo) {
  int idx = blockIdx.x * blockDim.x + threadIdx.x;
  if (idx >= CDIM * CDIM) return;
  oq[idx] = f2bf(Wq[idx]); ok[idx] = f2bf(Wk[idx]);
  ov[idx] = f2bf(Wv[idx]); oo[idx] = f2bf(Wo[idx]);
}

// ---------------- bf16 WMMA GEMM: out[4096][512] = X @ W^T + b --------------
// workgroup = 8 wave32 -> 128(M) x 64(N) tile; W-tile staged in LDS
// (double-buffered, shared by all 8 waves), A fragments pipelined from global.
// mode 0/1: Q/K bf16 [b][h][n][d];  mode 2: V*(1+g*sal) f32 [b][c][n];
// mode 3: final fp32 [b][c][h][w]
__global__ __launch_bounds__(256)
void gemm_bf16(const unsigned short* __restrict__ X,
               const unsigned short* __restrict__ Wb,
               const float* __restrict__ bias, int mode,
               unsigned short* __restrict__ outH,
               float* __restrict__ outF,
               const float* __restrict__ sal,
               const float* __restrict__ gammap) {
  __shared__ __align__(16) unsigned short Wsh[2][64][32];   // 8 KB
  int tid = threadIdx.x;
  int lane = tid & 31, wv = tid >> 5;
  int hlf = lane >> 4, ln = lane & 15;
  int mblk = blockIdx.x & 31;               // 32 M-blocks of 128
  int nt = blockIdx.x >> 5;                 // 8 N-supertiles of 64
  int j0 = nt * 64;
  int mrow = mblk * 128 + wv * 16;

  // cooperative W-tile copy mapping: 64 rows x 32 cols, 16B per thread
  int wr = tid >> 2, wc = (tid & 3) * 8;
  const unsigned short* wsrc = Wb + (j0 + wr) * CDIM + wc;
  unsigned short* wdst[2] = { &Wsh[0][0][0] + wr * 32 + wc,
                              &Wsh[1][0][0] + wr * 32 + wc };

  Acc acc[4];
#pragma unroll
  for (int t = 0; t < 4; ++t) {
    float bv = bias[j0 + t * 16 + ln];
#pragma unroll
    for (int r = 0; r < 8; ++r) acc[t].f[r] = bv;
  }

  Frag a_cur; load_frag(a_cur, X, CDIM, mrow + ln, 0, hlf);
#ifdef HAVE_ASYNC
  cp16_async(wsrc, wdst[0]);
#else
  uint4 wreg = *(const uint4*)(wsrc);
#endif

  for (int ks = 0; ks < 16; ++ks) {
    int cur = ks & 1;
    const unsigned short* wsh = &Wsh[cur][0][0];
#ifdef HAVE_ASYNC
    __builtin_amdgcn_s_wait_asynccnt(0);
    __syncthreads();
    if (ks < 15) cp16_async(wsrc + (ks + 1) * 32, wdst[cur ^ 1]);
#else
    *(uint4*)wdst[cur] = wreg;
    __syncthreads();
    if (ks < 15) wreg = *(const uint4*)(wsrc + (ks + 1) * 32);
#endif
    Frag a = a_cur;                                  // pipeline A from global
    if (ks < 15) load_frag(a_cur, X, CDIM, mrow + ln, (ks + 1) * 32, hlf);
#pragma unroll
    for (int t = 0; t < 4; ++t) {
      Frag b; load_frag(b, wsh, 32, t * 16 + ln, 0, hlf);   // ds_load_b128
      acc[t].v = __builtin_amdgcn_wmma_f32_16x16x32_bf16(
          false, a.v, false, b.v, (short)0, acc[t].v, false, false);
    }
  }

  float gam = (mode == 2) ? *gammap : 0.f;
#pragma unroll
  for (int t = 0; t < 4; ++t) {
    int j = j0 + t * 16 + ln;
#pragma unroll
    for (int r = 0; r < 8; ++r) {
      int row = mrow + r + 8 * hlf;         // 0..4095
      int b = row >> 10, q = row & (NQ - 1);
      float val = acc[t].f[r];
      if (mode <= 1) {                      // Q or K -> [b][h][q][d] bf16
        int h = j >> 6, d = j & 63;
        outH[(((b * NH + h) * NQ) + q) * HD + d] = f2bf(val);
      } else if (mode == 2) {               // V * (1 + gamma*sal) -> [b][c][q]
        float s = sal[b * NQ + q];
        outF[(b * CDIM + j) * NQ + q] = val * (1.f + gam * s);
      } else {                              // final out -> [b][c][h][w]
        outF[(b * CDIM + j) * NQ + q] = val;
      }
    }
  }
}

// ---------------- grid-sample coordinate precompute -------------------------
__global__ __launch_bounds__(256)
void coords_kernel(const float* __restrict__ off, float2* __restrict__ cw,
                   int4* __restrict__ ci) {
  int idx = blockIdx.x * blockDim.x + threadIdx.x;  // B*N
  if (idx >= BATCH * NQ) return;
  int b = idx >> 10, p = idx & (NQ - 1);
  int px = p & 31, py = p >> 5;
  float gx = -1.f + 2.f * (float)px / 31.f + off[(b * 2 + 0) * NQ + p] * (2.f / WID);
  float gy = -1.f + 2.f * (float)py / 31.f + off[(b * 2 + 1) * NQ + p] * (2.f / HGT);
  float x = fminf(fmaxf(((gx + 1.f) * WID - 1.f) * 0.5f, 0.f), WID - 1.f);
  float y = fminf(fmaxf(((gy + 1.f) * HGT - 1.f) * 0.5f, 0.f), HGT - 1.f);
  float x0 = floorf(x), y0 = floorf(y);
  cw[idx] = make_float2(x - x0, y - y0);
  int x0i = (int)x0, y0i = (int)y0;
  ci[idx] = make_int4(x0i, y0i, min(x0i + 1, WID - 1), min(y0i + 1, HGT - 1));
}

// bilinear sample Vm[b][c][:,:] -> Vd transposed bf16 [b][c][key]
__global__ __launch_bounds__(256)
void sample_kernel(const float* __restrict__ Vm, const float2* __restrict__ cw,
                   const int4* __restrict__ ci, unsigned short* __restrict__ Vdt) {
  int idx = blockIdx.x * blockDim.x + threadIdx.x;  // [b][c][p]
  if (idx >= BATCH * CDIM * NQ) return;
  int p = idx & (NQ - 1);
  int b = idx >> 19;
  float2 w = cw[b * NQ + p];
  int4  q = ci[b * NQ + p];
  const float* base = Vm + (idx & ~(NQ - 1));
  float v00 = base[q.y * WID + q.x];
  float v01 = base[q.y * WID + q.z];
  float v10 = base[q.w * WID + q.x];
  float v11 = base[q.w * WID + q.z];
  float val = v00 * (1.f - w.x) * (1.f - w.y) + v01 * w.x * (1.f - w.y)
            + v10 * (1.f - w.x) * w.y + v11 * w.x * w.y;
  Vdt[idx] = f2bf(val);
}

// ---------------- fused flash attention -------------------------------------
// grid: 32 (b,h) x 8 q-blocks of 128; block = 8 wave32, wave: 16q x hd 64.
// K-tile (64x64) and V-tile (64x64) staged in double-buffered LDS, shared by
// all 8 waves; next key-block's tiles fetched while computing current one.
__global__ __launch_bounds__(256)
void attn_kernel(const unsigned short* __restrict__ Qb,
                 const unsigned short* __restrict__ Kb,
                 const unsigned short* __restrict__ Vdt,
                 const float* __restrict__ Pth,
                 const float* __restrict__ lamp,
                 unsigned short* __restrict__ Obuf) {
  __shared__ __align__(16) unsigned short Ksh[2][64][64];   // 16 KB
  __shared__ __align__(16) unsigned short Vsh[2][64][64];   // 16 KB
  __shared__ __align__(16) unsigned short Pls[8][16][64];   // 16 KB per-wave P
  int tid = threadIdx.x;
  int lane = tid & 31, wv = tid >> 5;
  int hlf = lane >> 4, ln = lane & 15;
  int bh = blockIdx.x & 31;
  int qblk = blockIdx.x >> 5;
  int b = bh >> 3, h = bh & 7;
  int q0 = qblk * 128 + wv * 16;
  const float scale = 0.125f;               // hd^-0.5
  float lam = *lamp;

  const unsigned short* Qrow = Qb + (size_t)((b * NH + h) * NQ) * HD;
  const unsigned short* Krow = Kb + (size_t)((b * NH + h) * NQ) * HD;
  const unsigned short* Vrow = Vdt + (size_t)(b * CDIM + h * HD) * NQ;

  // cooperative tile copy mapping (4096 ushorts = 512 x b128 per tile)
  int i0 = tid, i1 = tid + 256;
  int vr0 = i0 >> 3, vc0 = (i0 & 7) * 8;
  int vr1 = i1 >> 3, vc1 = (i1 & 7) * 8;

  Frag qf[2];                               // Q fragments, reused all blocks
#pragma unroll
  for (int ks = 0; ks < 2; ++ks) load_frag(qf[ks], Qrow, HD, q0 + ln, ks * 32, hlf);

  float pb[8], mrun[8], lrun[8];
  Acc o[4];
#pragma unroll
  for (int r = 0; r < 8; ++r) {
    pb[r] = lam * Pth[b * NQ + q0 + r + 8 * hlf];
    mrun[r] = -1e30f; lrun[r] = 0.f;
  }
#pragma unroll
  for (int t = 0; t < 4; ++t)
#pragma unroll
    for (int r = 0; r < 8; ++r) o[t].f[r] = 0.f;

#ifdef HAVE_ASYNC
  {
    cp16_async(Krow + i0 * 8, &Ksh[0][0][0] + i0 * 8);
    cp16_async(Krow + i1 * 8, &Ksh[0][0][0] + i1 * 8);
    cp16_async(Vrow + vr0 * NQ + vc0, &Vsh[0][0][0] + i0 * 8);
    cp16_async(Vrow + vr1 * NQ + vc1, &Vsh[0][0][0] + i1 * 8);
  }
#else
  uint4 kreg0 = ((const uint4*)Krow)[i0];
  uint4 kreg1 = ((const uint4*)Krow)[i1];
  uint4 vreg0 = *(const uint4*)(Vrow + vr0 * NQ + vc0);
  uint4 vreg1 = *(const uint4*)(Vrow + vr1 * NQ + vc1);
#endif

  for (int kb = 0; kb < NQ / 64; ++kb) {
    int cur = kb & 1;
    const unsigned short* ksh = &Ksh[cur][0][0];
    const unsigned short* vsh = &Vsh[cur][0][0];
#ifdef HAVE_ASYNC
    __builtin_amdgcn_s_wait_asynccnt(0);
    __syncthreads();
    if (kb < 15) {
      const unsigned short* kg = Krow + (kb + 1) * 64 * HD;
      const unsigned short* vg = Vrow + (kb + 1) * 64;
      unsigned short* kd = &Ksh[cur ^ 1][0][0];
      unsigned short* vd = &Vsh[cur ^ 1][0][0];
      cp16_async(kg + i0 * 8, kd + i0 * 8);
      cp16_async(kg + i1 * 8, kd + i1 * 8);
      cp16_async(vg + vr0 * NQ + vc0, vd + i0 * 8);
      cp16_async(vg + vr1 * NQ + vc1, vd + i1 * 8);
    }
#else
    {
      unsigned short* kd = &Ksh[cur][0][0];
      unsigned short* vd = &Vsh[cur][0][0];
      ((uint4*)kd)[i0] = kreg0; ((uint4*)kd)[i1] = kreg1;
      ((uint4*)vd)[i0] = vreg0; ((uint4*)vd)[i1] = vreg1;
    }
    __syncthreads();
    if (kb < 15) {
      const unsigned short* kg = Krow + (kb + 1) * 64 * HD;
      const unsigned short* vg = Vrow + (kb + 1) * 64;
      kreg0 = ((const uint4*)kg)[i0];
      kreg1 = ((const uint4*)kg)[i1];
      vreg0 = *(const uint4*)(vg + vr0 * NQ + vc0);
      vreg1 = *(const uint4*)(vg + vr1 * NQ + vc1);
    }
#endif
    if (kb < 14) {                          // L2 prefetch two blocks ahead
      __builtin_prefetch(Krow + (kb + 2) * 64 * HD, 0, 1);
      __builtin_prefetch(Vrow + (kb + 2) * 64, 0, 1);
    }

    Acc s[4];
#pragma unroll
    for (int t = 0; t < 4; ++t)
#pragma unroll
      for (int r = 0; r < 8; ++r) s[t].f[r] = 0.f;

#pragma unroll
    for (int ks = 0; ks < 2; ++ks)
#pragma unroll
      for (int t = 0; t < 4; ++t) {
        Frag kf; load_frag(kf, ksh, 64, t * 16 + ln, ks * 32, hlf);
        s[t].v = __builtin_amdgcn_wmma_f32_16x16x32_bf16(
            false, qf[ks].v, false, kf.v, (short)0, s[t].v, false, false);
      }

    // ---- online softmax (rows live across the 16 lanes of each half) ----
    float rm[8];
#pragma unroll
    for (int r = 0; r < 8; ++r) {
      float v = -1e30f;
#pragma unroll
      for (int t = 0; t < 4; ++t) { s[t].f[r] = s[t].f[r] * scale + pb[r]; v = fmaxf(v, s[t].f[r]); }
      rm[r] = v;
    }
#pragma unroll
    for (int off = 1; off < 16; off <<= 1)
#pragma unroll
      for (int r = 0; r < 8; ++r) rm[r] = fmaxf(rm[r], __shfl_xor(rm[r], off, 32));

    float rs[8];
#pragma unroll
    for (int r = 0; r < 8; ++r) {
      float mn = fmaxf(mrun[r], rm[r]);
      float al = __expf(mrun[r] - mn);
      mrun[r] = mn;
      float sum = 0.f;
#pragma unroll
      for (int t = 0; t < 4; ++t) {
        float pv = __expf(s[t].f[r] - mn);
        s[t].f[r] = pv;
        sum += pv;
      }
      rs[r] = sum;
      lrun[r] *= al;
#pragma unroll
      for (int t = 0; t < 4; ++t) o[t].f[r] *= al;
    }
#pragma unroll
    for (int off = 1; off < 16; off <<= 1)
#pragma unroll
      for (int r = 0; r < 8; ++r) rs[r] += __shfl_xor(rs[r], off, 32);
#pragma unroll
    for (int r = 0; r < 8; ++r) lrun[r] += rs[r];

    // stage P (bf16) in this wave's LDS region, re-read as A fragments
#pragma unroll
    for (int t = 0; t < 4; ++t)
#pragma unroll
      for (int r = 0; r < 8; ++r)
        Pls[wv][r + 8 * hlf][t * 16 + ln] = f2bf(s[t].f[r]);

#pragma unroll
    for (int ks = 0; ks < 2; ++ks) {
      Frag pa; load_frag(pa, &Pls[wv][0][0], 64, ln, ks * 32, hlf);
#pragma unroll
      for (int t = 0; t < 4; ++t) {
        Frag vb; load_frag(vb, vsh, 64, t * 16 + ln, ks * 32, hlf);
        o[t].v = __builtin_amdgcn_wmma_f32_16x16x32_bf16(
            false, pa.v, false, vb.v, (short)0, o[t].v, false, false);
      }
    }
  }

  // normalize and store O as bf16 [b*NQ + q][c] for the output projection
#pragma unroll
  for (int r = 0; r < 8; ++r) {
    float inv = 1.f / lrun[r];
    int qi = q0 + r + 8 * hlf;
#pragma unroll
    for (int t = 0; t < 4; ++t) {
      int c = h * HD + t * 16 + ln;
      Obuf[(size_t)(b * NQ + qi) * CDIM + c] = f2bf(o[t].f[r] * inv);
    }
  }
}

// ---------------- host orchestration ----------------
extern "C" void kernel_launch(void* const* d_in, const int* in_sizes, int n_in,
                              void* d_out, int out_size, void* d_ws, size_t ws_size,
                              hipStream_t stream) {
  (void)in_sizes; (void)n_in; (void)out_size; (void)ws_size;
  const float* q_feat  = (const float*)d_in[0];
  const float* kv_feat = (const float*)d_in[1];
  const float* offsets = (const float*)d_in[2];
  const float* sal     = (const float*)d_in[3];
  const float* pth     = (const float*)d_in[4];
  const float* Wq = (const float*)d_in[5];   const float* bq = (const float*)d_in[6];
  const float* Wk = (const float*)d_in[7];   const float* bk = (const float*)d_in[8];
  const float* Wv = (const float*)d_in[9];   const float* bv = (const float*)d_in[10];
  const float* Wo = (const float*)d_in[11];  const float* bo = (const float*)d_in[12];
  const float* lamp = (const float*)d_in[13];
  const float* gam  = (const float*)d_in[14];

  char* ws = (char*)d_ws;
  unsigned short* Xq  = (unsigned short*)(ws);                    // 4 MB [b][n][c]
  unsigned short* Xk  = (unsigned short*)(ws + (4ull  << 20));    // 4 MB
  unsigned short* Wqb = (unsigned short*)(ws + (8ull  << 20));    // 4 x 0.5 MB
  unsigned short* Wkb = Wqb + CDIM * CDIM;
  unsigned short* Wvb = Wkb + CDIM * CDIM;
  unsigned short* Wob = Wvb + CDIM * CDIM;
  unsigned short* Qb  = (unsigned short*)(ws + (10ull << 20));    // 4 MB [b][h][n][d]
  unsigned short* Kb  = (unsigned short*)(ws + (14ull << 20));    // 4 MB
  float*          Vm  = (float*)         (ws + (18ull << 20));    // 8 MB [b][c][n]
  unsigned short* Vdt = (unsigned short*)(ws + (26ull << 20));    // 4 MB [b][c][key]
  unsigned short* Ob  = (unsigned short*)(ws + (30ull << 20));    // 4 MB [b*n][c]
  float2*         cw  = (float2*)        (ws + (34ull << 20));
  int4*           ci  = (int4*)          (ws + (34ull << 20) + (1u << 18));

  float* out = (float*)d_out;

  pack_w<<<(CDIM * CDIM + 255) / 256, 256, 0, stream>>>(Wq, Wk, Wv, Wo, Wqb, Wkb, Wvb, Wob);
  pack_feat<<<(BATCH * NQ * CDIM + 255) / 256, 256, 0, stream>>>(q_feat, kv_feat, Xq, Xk);

  dim3 gemmGrid(32 * 8);   // 32 M-blocks(128) x 8 N-supertiles(64), 8 waves each
  gemm_bf16<<<gemmGrid, 256, 0, stream>>>(Xq, Wqb, bq, 0, Qb, nullptr, nullptr, nullptr);
  gemm_bf16<<<gemmGrid, 256, 0, stream>>>(Xk, Wkb, bk, 1, Kb, nullptr, nullptr, nullptr);
  gemm_bf16<<<gemmGrid, 256, 0, stream>>>(Xk, Wvb, bv, 2, nullptr, Vm, sal, gam);

  coords_kernel<<<(BATCH * NQ + 255) / 256, 256, 0, stream>>>(offsets, cw, ci);
  sample_kernel<<<(BATCH * CDIM * NQ + 255) / 256, 256, 0, stream>>>(Vm, cw, ci, Vdt);

  attn_kernel<<<32 * 8, 256, 0, stream>>>(Qb, Kb, Vdt, pth, lamp, Ob);

  gemm_bf16<<<gemmGrid, 256, 0, stream>>>(Ob, Wob, bo, 3, nullptr, out, nullptr, nullptr);
}
